// LongShortModel_76209899700577
// MI455X (gfx1250) — matrix operand order repaired and checked by reading, not compiled, not run on hardware
//
#include <hip/hip_runtime.h>
#include <hip/hip_bf16.h>

typedef __attribute__((ext_vector_type(16))) _Float16 v16h;
typedef __attribute__((ext_vector_type(8)))  float    v8f;

#define NN    50000
#define EE    800000
#define ETOT  850000   // EE + NN self loops
#define HC    128      // H*C
#define KDIM  128

// ---------------------------------------------------------------------------
// monotonic-uint encoding for float atomic max
// ---------------------------------------------------------------------------
__device__ __forceinline__ unsigned enc_f32(float f) {
  unsigned u = __float_as_uint(f);
  return (u & 0x80000000u) ? ~u : (u | 0x80000000u);
}
__device__ __forceinline__ float dec_f32(unsigned u) {
  return (u & 0x80000000u) ? __uint_as_float(u & 0x7FFFFFFFu)
                           : __uint_as_float(~u);
}
__device__ __forceinline__ float lrelu(float v, float s) {
  return v > 0.f ? v : v * s;
}

// ---------------------------------------------------------------------------
// W (fp32 [128,128]) -> f16 copy, once per layer
// ---------------------------------------------------------------------------
__global__ void convert_w16(const float* __restrict__ W,
                            _Float16* __restrict__ Wh) {
  int i = blockIdx.x * blockDim.x + threadIdx.x;
  if (i < KDIM * HC) Wh[i] = (_Float16)W[i];
}

// ---------------------------------------------------------------------------
// WMMA GEMM: Hout[N,128](f16) = X[N,128](f32) @ Wh[128,128](f16)
// one wave -> one 16x64 strip: 4 independent accumulators sharing one A
// fragment (4x A reuse, 4 parallel WMMA dep-chains to fill the XDL pipe).
// B fragments load directly as v16h; D stored as f16 (feeds the f16 gather).
// block = 64 threads = 2 waves -> full 16x128 row strip; grid = N/16.
// ---------------------------------------------------------------------------
__global__ __launch_bounds__(64)
void gat_gemm_wmma(const float* __restrict__ X, const _Float16* __restrict__ Wh,
                   _Float16* __restrict__ Hout) {
  const int wave   = threadIdx.x >> 5;   // 0..1
  const int lane   = threadIdx.x & 31;
  const int tile_m = blockIdx.x;
  const int hi     = lane >> 4;          // 0 for lanes 0-15, 1 for 16-31

  const int row_a  = tile_m * 16 + (lane & 15);   // A: lane = M row
  const int ka     = hi * 8;                      // A: K sub-base within chunk
  const int kb_row = (lane & 15) + hi * 16;       // unused; B lane -> N column (see wb)
  const int col0   = wave * 64;                   // wave covers cols [col0, col0+64)

  v8f acc0 = {}, acc1 = {}, acc2 = {}, acc3 = {};
#pragma unroll
  for (int k0 = 0; k0 < KDIM; k0 += 32) {
    v16h a;
    const float* xa = X + (size_t)row_a * KDIM + k0 + ka;
#pragma unroll
    for (int j = 0; j < 8; ++j) {
      a[j]     = (_Float16)xa[j];        // K = k0+ka   .. +7
      a[8 + j] = (_Float16)xa[16 + j];   // K = k0+ka+16.. +23
    }
    const _Float16* wb = Wh + (size_t)(k0 + ka) * HC + col0 + (lane & 15);
    v16h b0, b1, b2, b3;
#pragma unroll
    for (int j = 0; j < 8; ++j) {
      b0[j]     = wb[(size_t)j * HC];
      b0[8 + j] = wb[(size_t)(16 + j) * HC];
      b1[j]     = wb[(size_t)j * HC + 16];
      b1[8 + j] = wb[(size_t)(16 + j) * HC + 16];
      b2[j]     = wb[(size_t)j * HC + 32];
      b2[8 + j] = wb[(size_t)(16 + j) * HC + 32];
      b3[j]     = wb[(size_t)j * HC + 48];
      b3[8 + j] = wb[(size_t)(16 + j) * HC + 48];
    }

    acc0 = __builtin_amdgcn_wmma_f32_16x16x32_f16(false, a, false, b0,
                                                  (short)0, acc0, false, false);
    acc1 = __builtin_amdgcn_wmma_f32_16x16x32_f16(false, a, false, b1,
                                                  (short)0, acc1, false, false);
    acc2 = __builtin_amdgcn_wmma_f32_16x16x32_f16(false, a, false, b2,
                                                  (short)0, acc2, false, false);
    acc3 = __builtin_amdgcn_wmma_f32_16x16x32_f16(false, a, false, b3,
                                                  (short)0, acc3, false, false);
  }

  const int row_d  = tile_m * 16 + hi * 8;  // D: VGPR r -> M = r + 8*hi
  const int lane_n = lane & 15;             // D: lane -> N
#pragma unroll
  for (int r = 0; r < 8; ++r) {
    _Float16* o = Hout + (size_t)(row_d + r) * HC + col0 + lane_n;
    o[0]  = (_Float16)acc0[r];
    o[16] = (_Float16)acc1[r];
    o[32] = (_Float16)acc2[r];
    o[48] = (_Float16)acc3[r];
  }
}

// ---------------------------------------------------------------------------
// per-node attention logits: att_s[n,h] = sum_c h[n,h,c]*a_src[h,c]
// block = 128 threads (4 waves), one node per block, one head per wave
// ---------------------------------------------------------------------------
__global__ __launch_bounds__(128)
void gat_att(const _Float16* __restrict__ Hh, const float* __restrict__ a_src,
             const float* __restrict__ a_dst, float* __restrict__ att_s,
             float* __restrict__ att_d) {
  const int n = blockIdx.x;
  const int t = threadIdx.x;
  float hv = (float)Hh[(size_t)n * HC + t];
  float vs = hv * a_src[t];     // a_src is [H,C] flat = 128 values
  float vd = hv * a_dst[t];
#pragma unroll
  for (int off = 16; off > 0; off >>= 1) {
    vs += __shfl_down(vs, off, 32);
    vd += __shfl_down(vd, off, 32);
  }
  if ((t & 31) == 0) {
    att_s[n * 4 + (t >> 5)] = vs;
    att_d[n * 4 + (t >> 5)] = vd;
  }
}

// ---------------------------------------------------------------------------
__global__ void zero_f32(float* __restrict__ p, int n) {
  int i = blockIdx.x * blockDim.x + threadIdx.x;
  if (i < n) p[i] = 0.0f;
}

// ---------------------------------------------------------------------------
// edge pass 1: segment max of leaky-relu logits (uint-encoded atomic max)
// one thread per (edge, head); self-loop edges e >= EE have src=dst=e-EE
// ---------------------------------------------------------------------------
__global__ void edge_max(const int* __restrict__ ei,
                         const float* __restrict__ att_s,
                         const float* __restrict__ att_d,
                         unsigned* __restrict__ m) {
  int idx = blockIdx.x * blockDim.x + threadIdx.x;
  if (idx >= ETOT * 4) return;
  int e = idx >> 2, hh = idx & 3;
  int src, dst;
  if (e < EE) { src = ei[e]; dst = ei[EE + e]; } else { src = dst = e - EE; }
  float v = lrelu(att_s[src * 4 + hh] + att_d[dst * 4 + hh], 0.2f);
  atomicMax(&m[dst * 4 + hh], enc_f32(v));
}

// ---------------------------------------------------------------------------
// edge pass 2: w = exp(e - m); z[dst] += w; agg[dst] += w * h[src]
// one block (128 threads = 4 waves) per edge; one head per wave.
// h gathered in f16 (halves the dominant HBM traffic); agg stays fp32.
// ---------------------------------------------------------------------------
__global__ __launch_bounds__(128)
void edge_agg(const int* __restrict__ ei, const float* __restrict__ att_s,
              const float* __restrict__ att_d, const unsigned* __restrict__ m,
              const _Float16* __restrict__ Hh, float* __restrict__ z,
              float* __restrict__ agg) {
  const int e = blockIdx.x;
  const int t = threadIdx.x;
  const int hh = t >> 5;
  int src, dst;
  if (e < EE) { src = ei[e]; dst = ei[EE + e]; } else { src = dst = e - EE; }
  float v = lrelu(att_s[src * 4 + hh] + att_d[dst * 4 + hh], 0.2f);
  float w = __expf(v - dec_f32(m[dst * 4 + hh]));
  if ((t & 31) == 0) atomicAdd(&z[dst * 4 + hh], w);
  float hval = (float)Hh[(size_t)src * HC + t];
  atomicAdd(&agg[(size_t)dst * HC + t], w * hval);
}

// ---------------------------------------------------------------------------
// per-node normalize: agg/(z+eps) + bias, then leaky-relu(0.01); in place
// ---------------------------------------------------------------------------
__global__ __launch_bounds__(128)
void node_norm(float* __restrict__ agg, const float* __restrict__ z,
               const float* __restrict__ bias) {
  const int n = blockIdx.x;
  const int t = threadIdx.x;
  float v = agg[(size_t)n * HC + t] / (z[n * 4 + (t >> 5)] + 1e-16f) + bias[t];
  agg[(size_t)n * HC + t] = lrelu(v, 0.01f);
}

// ---------------------------------------------------------------------------
// final FC: out[n] = h[n,:] . Wfc + bfc   (reads fp32 normalized layer-2 out)
// ---------------------------------------------------------------------------
__global__ __launch_bounds__(128)
void final_fc(const float* __restrict__ Hf, const float* __restrict__ Wfc,
              const float* __restrict__ bfc, float* __restrict__ out) {
  __shared__ float red[4];
  const int n = blockIdx.x;
  const int t = threadIdx.x;
  float v = Hf[(size_t)n * HC + t] * Wfc[t];
#pragma unroll
  for (int off = 16; off > 0; off >>= 1) v += __shfl_down(v, off, 32);
  if ((t & 31) == 0) red[t >> 5] = v;
  __syncthreads();
  if (t == 0) out[n] = red[0] + red[1] + red[2] + red[3] + bfc[0];
}

// ---------------------------------------------------------------------------
extern "C" void kernel_launch(void* const* d_in, const int* in_sizes, int n_in,
                              void* d_out, int out_size, void* d_ws, size_t ws_size,
                              hipStream_t stream) {
  const float* x   = (const float*)d_in[0];
  const int*   ei  = (const int*)  d_in[1];
  const float* W1  = (const float*)d_in[2];
  const float* as1 = (const float*)d_in[3];
  const float* ad1 = (const float*)d_in[4];
  const float* b1  = (const float*)d_in[5];
  const float* W2  = (const float*)d_in[6];
  const float* as2 = (const float*)d_in[7];
  const float* ad2 = (const float*)d_in[8];
  const float* b2  = (const float*)d_in[9];
  const float* Wfc = (const float*)d_in[10];
  const float* bfc = (const float*)d_in[11];
  float* out = (float*)d_out;

  // workspace layout: f16 regions first (alignment), then fp32
  _Float16* Wh   = (_Float16*)d_ws;                          // [128,128] f16
  _Float16* hA   = Wh + (size_t)KDIM * HC;                   // [N,128]   f16 h
  float*    bufB = (float*)(hA + (size_t)NN * HC);           // [N,128]   f32 agg
  float*    att_s = bufB + (size_t)NN * HC;                  // [N,4]
  float*    att_d = att_s + (size_t)NN * 4;                  // [N,4]
  unsigned* mbuf  = (unsigned*)(att_d + (size_t)NN * 4);     // [N,4] encoded max
  float*    zbuf  = (float*)(mbuf + (size_t)NN * 4);         // [N,4]

  const int gemm_blocks = NN / 16;   // 3125 blocks x 64 threads (2 waves)
  const int nzB  = NN * HC;          // agg zero count
  const int nzMZ = NN * 8;           // m + z contiguous zero count
  const int edge_threads = ETOT * 4;
  const int wconv_blocks = (KDIM * HC + 255) / 256;

  // ---------------- layer 1 ----------------
  convert_w16<<<wconv_blocks, 256, 0, stream>>>(W1, Wh);
  gat_gemm_wmma<<<gemm_blocks, 64, 0, stream>>>(x, Wh, hA);
  gat_att<<<NN, 128, 0, stream>>>(hA, as1, ad1, att_s, att_d);
  zero_f32<<<(nzB + 255) / 256, 256, 0, stream>>>(bufB, nzB);
  zero_f32<<<(nzMZ + 255) / 256, 256, 0, stream>>>((float*)mbuf, nzMZ);
  edge_max<<<(edge_threads + 255) / 256, 256, 0, stream>>>(ei, att_s, att_d, mbuf);
  edge_agg<<<ETOT, 128, 0, stream>>>(ei, att_s, att_d, mbuf, hA, zbuf, bufB);
  node_norm<<<NN, 128, 0, stream>>>(bufB, zbuf, b1);

  // ---------------- layer 2 ----------------
  convert_w16<<<wconv_blocks, 256, 0, stream>>>(W2, Wh);
  gat_gemm_wmma<<<gemm_blocks, 64, 0, stream>>>(bufB, Wh, hA);
  gat_att<<<NN, 128, 0, stream>>>(hA, as2, ad2, att_s, att_d);
  zero_f32<<<(nzB + 255) / 256, 256, 0, stream>>>(bufB, nzB);   // safe: gemm consumed bufB
  zero_f32<<<(nzMZ + 255) / 256, 256, 0, stream>>>((float*)mbuf, nzMZ);
  edge_max<<<(edge_threads + 255) / 256, 256, 0, stream>>>(ei, att_s, att_d, mbuf);
  edge_agg<<<ETOT, 128, 0, stream>>>(ei, att_s, att_d, mbuf, hA, zbuf, bufB);
  node_norm<<<NN, 128, 0, stream>>>(bufB, zbuf, b2);

  // ---------------- head ----------------
  final_fc<<<NN, 128, 0, stream>>>(bufB, Wfc, bfc, out);
}
